// GATModel_19473381720258
// MI455X (gfx1250) — matrix operand order repaired and checked by reading, not compiled, not run on hardware
//
#include <hip/hip_runtime.h>
#include <hip/hip_bf16.h>
#include <stdint.h>

// ---------------------------------------------------------------------------
// GAT 3-layer pipeline for MI455X (gfx1250, wave32, WMMA).
// fp32 operands are pre-packed per layer into bf16-pair dwords in this code's
// v_wmma_f32_16x16x32_bf16 fragment layouts (contiguous k = 16*half + s per lane):
//   Xp[row][k/2]            -> A fragments load as 2x global_load_b128
//   Wp[tn][kg][lane][8]     -> B fragments load as 2x global_load_b128
// GEMM is templated on (K, Mout) so all loop offsets are immediates; one wave
// computes a 16 x Mout strip (A reused across Mout/16 WMMAs per k-step).
// ---------------------------------------------------------------------------

#define GN      100000
#define GIN     256
#define GHID    128
#define GOUT    64
#define GHEADS  4
#define NEG_SLOPE 0.2f

typedef __attribute__((ext_vector_type(16))) __bf16 v16bf;
typedef __attribute__((ext_vector_type(8)))  float  v8f;

__device__ __forceinline__ unsigned pack_bf16(float a, float b) {
  unsigned ua = __float_as_uint(a);
  unsigned ub = __float_as_uint(b);
  ua += 0x7FFFu + ((ua >> 16) & 1u);   // round-to-nearest-even bf16
  ub += 0x7FFFu + ((ub >> 16) & 1u);
  return (ua >> 16) | (ub & 0xFFFF0000u);
}

// X[R,K] fp32 row-major -> Xp[R, K/2] u32, Xp[r][j] = pack(X[r,2j], X[r,2j+1])
__global__ void pack_rows(const float* __restrict__ X, unsigned* __restrict__ Xp,
                          int R, int K) {
  const long long i = (long long)blockIdx.x * blockDim.x + threadIdx.x;
  if (i >= (long long)R * (K >> 1)) return;
  const float2 v = *(const float2*)(X + 2 * i);
  Xp[i] = pack_bf16(v.x, v.y);
}

// W[K,M] fp32 row-major -> fragment-contiguous Wp[tn][kg][lane][q]:
//   k = kg*32 + (lane>>4)*16 + 2q, col = tn*16 + (lane&15)
__global__ void pack_wfrag(const float* __restrict__ W, unsigned* __restrict__ Wp,
                           int K, int M) {
  const int i = blockIdx.x * blockDim.x + threadIdx.x;
  const int KG = K >> 5;
  if (i >= (M >> 4) * KG * 256) return;
  const int q    = i & 7;
  const int lane = (i >> 3) & 31;
  const int kg   = (i >> 8) % KG;
  const int tn   = (i >> 8) / KG;
  const int k    = kg * 32 + ((lane >> 4) << 4) + 2 * q;
  const int col  = tn * 16 + (lane & 15);
  Wp[i] = pack_bf16(W[(size_t)k * M + col], W[(size_t)(k + 1) * M + col]);
}

// Y[n_rows x Mout] = X @ W. One wave per 16 x Mout strip, K fully unrolled.
template <int K, int Mout>
__global__ void gemm_wmma_bf16(const unsigned* __restrict__ Xp,
                               const unsigned* __restrict__ Wp,
                               float* __restrict__ Y, int n_rows) {
  constexpr int NT = Mout / 16;        // n-tiles (accumulators) per wave
  constexpr int KG = K / 32;           // k-steps
  const int lane = threadIdx.x & 31;
  const int tm   = blockIdx.x * (blockDim.x >> 5) + (threadIdx.x >> 5);
  if (tm >= (n_rows >> 4)) return;     // uniform per-wave: EXEC stays full
  const int half = lane >> 4;
  const int l16  = lane & 15;

  const unsigned* __restrict__ xprow = Xp + (size_t)(tm * 16 + l16) * (K / 2);
  const unsigned* __restrict__ wlane = Wp + (size_t)lane * 8;

  v8f acc[NT];
#pragma unroll
  for (int t = 0; t < NT; ++t) acc[t] = v8f{};

#pragma unroll
  for (int kg = 0; kg < KG; ++kg) {
    union { v16bf v; unsigned u[8]; } A;
#pragma unroll
    for (int i = 0; i < 8; ++i)        // 2x b128: fully contiguous
      A.u[i] = xprow[kg * 16 + half * 8 + i];
#pragma unroll
    for (int t = 0; t < NT; ++t) {
      union { v16bf v; unsigned u[8]; } B;
      const unsigned* __restrict__ bp = wlane + (size_t)(t * KG + kg) * 256;
#pragma unroll
      for (int i = 0; i < 8; ++i)      // 2x b128: fully contiguous
        B.u[i] = bp[i];
      acc[t] = __builtin_amdgcn_wmma_f32_16x16x32_bf16(
          false, A.v, false, B.v, (short)0, acc[t], false, false);
    }
  }
  float* __restrict__ yp = Y + (size_t)(tm * 16 + half * 8) * Mout + l16;
#pragma unroll
  for (int t = 0; t < NT; ++t)
#pragma unroll
    for (int j = 0; j < 8; ++j)        // D f32: VGPR j -> row j + 8*half
      yp[j * Mout + t * 16] = acc[t][j];
}

// alpha_s[n,h] = <h[n,h,:], a_src[h,:]>, alpha_d likewise
__global__ void alpha_kernel(const float* __restrict__ H,
                             const float* __restrict__ a_s,
                             const float* __restrict__ a_d,
                             float* __restrict__ out_s,
                             float* __restrict__ out_d,
                             int n, int heads, int C) {
  const int idx = blockIdx.x * blockDim.x + threadIdx.x;
  if (idx >= n * heads) return;
  const int node = idx / heads, h = idx % heads;
  const float* hp = H + (size_t)node * heads * C + (size_t)h * C;
  float ss = 0.f, sd = 0.f;
  for (int c = 0; c < C; ++c) {
    const float v = hp[c];
    ss += v * a_s[h * C + c];
    sd += v * a_d[h * C + c];
  }
  out_s[idx] = ss;
  out_d[idx] = sd;
}

__global__ void fill_u32(unsigned* __restrict__ p, unsigned v, size_t n) {
  const size_t i = (size_t)blockIdx.x * blockDim.x + threadIdx.x;
  if (i < n) p[i] = v;
}

__device__ __forceinline__ void edge_nodes(int e, int E,
                                           const int* __restrict__ src,
                                           const int* __restrict__ dst,
                                           int& s, int& d) {
  if (e < E) { s = src[e]; d = dst[e]; }
  else       { s = e - E; d = e - E; }     // self-loop
}
__device__ __forceinline__ float leaky(float x) {
  return x > 0.f ? x : NEG_SLOPE * x;
}
// monotonic uint encoding of float for atomicMax (init 0 == minimum)
__device__ __forceinline__ unsigned fenc(float f) {
  unsigned u = __float_as_uint(f);
  return u ^ ((unsigned)((int)u >> 31) | 0x80000000u);
}
__device__ __forceinline__ float fdec(unsigned k) {
  unsigned u = (k & 0x80000000u) ? (k ^ 0x80000000u) : ~k;
  return __uint_as_float(u);
}

__global__ void edge_max(const int* __restrict__ src, const int* __restrict__ dst,
                         const float* __restrict__ as_, const float* __restrict__ ad_,
                         unsigned* __restrict__ mx, int E, int ET, int H) {
  const int e = blockIdx.x * blockDim.x + threadIdx.x;
  if (e >= ET) return;
  int s, d; edge_nodes(e, E, src, dst, s, d);
  for (int h = 0; h < H; ++h) {
    const float v = leaky(as_[s * H + h] + ad_[d * H + h]);
    atomicMax(&mx[d * H + h], fenc(v));
  }
}

__global__ void edge_sum(const int* __restrict__ src, const int* __restrict__ dst,
                         const float* __restrict__ as_, const float* __restrict__ ad_,
                         const unsigned* __restrict__ mx, float* __restrict__ denom,
                         int E, int ET, int H) {
  const int e = blockIdx.x * blockDim.x + threadIdx.x;
  if (e >= ET) return;
  int s, d; edge_nodes(e, E, src, dst, s, d);
  for (int h = 0; h < H; ++h) {
    const float v = leaky(as_[s * H + h] + ad_[d * H + h]);
    const float m = fdec(mx[d * H + h]);
    atomicAdd(&denom[d * H + h], __expf(v - m));
  }
}

// one block per edge, one thread per output channel (H*C threads)
__global__ void edge_agg(const int* __restrict__ src, const int* __restrict__ dst,
                         const float* __restrict__ as_, const float* __restrict__ ad_,
                         const unsigned* __restrict__ mx, const float* __restrict__ denom,
                         const float* __restrict__ Hbuf, float* __restrict__ agg,
                         int E, int ET, int H, int C) {
  const int e = blockIdx.x;
  if (e >= ET) return;
  const int c = threadIdx.x;               // 0 .. H*C-1
  const int h = c / C;
  int s, d; edge_nodes(e, E, src, dst, s, d);
  const float v     = leaky(as_[s * H + h] + ad_[d * H + h]);
  const float m     = fdec(mx[d * H + h]);
  const float alpha = __expf(v - m) / (denom[d * H + h] + 1e-16f);
  const int CH = H * C;
  atomicAdd(&agg[(size_t)d * CH + c], alpha * Hbuf[(size_t)s * CH + c]);
}

__global__ void node_finish(const float* __restrict__ agg,
                            const float* __restrict__ bias,
                            float* __restrict__ out, int n, int CH, int do_elu) {
  const int i = blockIdx.x * blockDim.x + threadIdx.x;
  if (i >= n * CH) return;
  float v = agg[i] + bias[i % CH];
  if (do_elu) v = v > 0.f ? v : (__expf(v) - 1.f);
  out[i] = v;
}

// ---------------------------------------------------------------------------

static inline int icdiv(long long a, long long b) { return (int)((a + b - 1) / b); }

struct LayerBufs {
  float *hbuf, *aggbuf;
  float *asb, *adb, *denb;
  unsigned *mxb, *xpk, *wpk;
  const int *src, *dst;
  int E, ET;
};

template <int K, int Mout>
static void run_gat_layer(const float* xin, const float* W,
                          const float* a_s, const float* a_d, const float* bias,
                          int H, float* outbuf, const LayerBufs& B,
                          int do_elu, hipStream_t stream) {
  const int C = Mout / H;
  // 0) pack fp32 -> bf16-pair dwords in fragment layouts
  pack_rows<<<icdiv((long long)GN * (K / 2), 256), 256, 0, stream>>>(
      xin, B.xpk, GN, K);
  pack_wfrag<<<icdiv((long long)(K / 2) * Mout, 256), 256, 0, stream>>>(
      W, B.wpk, K, Mout);
  // 1) GEMM via WMMA: one wave per 16 x Mout strip
  gemm_wmma_bf16<K, Mout><<<icdiv((long long)(GN / 16) * 32, 256), 256, 0, stream>>>(
      B.xpk, B.wpk, B.hbuf, GN);
  // 2) per-node attention logits
  alpha_kernel<<<icdiv((long long)GN * H, 256), 256, 0, stream>>>(
      B.hbuf, a_s, a_d, B.asb, B.adb, GN, H, C);
  // 3) zero max / denom / agg
  fill_u32<<<icdiv((long long)GN * H, 256), 256, 0, stream>>>(B.mxb, 0u, (size_t)GN * H);
  fill_u32<<<icdiv((long long)GN * H, 256), 256, 0, stream>>>((unsigned*)B.denb, 0u, (size_t)GN * H);
  fill_u32<<<icdiv((long long)GN * Mout, 256), 256, 0, stream>>>((unsigned*)B.aggbuf, 0u, (size_t)GN * Mout);
  // 4) segment max, exp-sum, weighted scatter
  edge_max<<<icdiv(B.ET, 256), 256, 0, stream>>>(B.src, B.dst, B.asb, B.adb, B.mxb, B.E, B.ET, H);
  edge_sum<<<icdiv(B.ET, 256), 256, 0, stream>>>(B.src, B.dst, B.asb, B.adb, B.mxb, B.denb, B.E, B.ET, H);
  edge_agg<<<B.ET, Mout, 0, stream>>>(B.src, B.dst, B.asb, B.adb, B.mxb, B.denb,
                                      B.hbuf, B.aggbuf, B.E, B.ET, H, C);
  // 5) bias (+ELU)
  node_finish<<<icdiv((long long)GN * Mout, 256), 256, 0, stream>>>(
      B.aggbuf, bias, outbuf, GN, Mout, do_elu);
}

extern "C" void kernel_launch(void* const* d_in, const int* in_sizes, int n_in,
                              void* d_out, int out_size, void* d_ws, size_t ws_size,
                              hipStream_t stream) {
  // inputs: x, edge_index, W1, as1, ad1, b1, W2, as2, ad2, b2, W3, as3, ad3, b3
  const float* x   = (const float*)d_in[0];
  const int* ei    = (const int*)d_in[1];
  const float* W1  = (const float*)d_in[2];
  const float* as1 = (const float*)d_in[3];
  const float* ad1 = (const float*)d_in[4];
  const float* b1  = (const float*)d_in[5];
  const float* W2  = (const float*)d_in[6];
  const float* as2 = (const float*)d_in[7];
  const float* ad2 = (const float*)d_in[8];
  const float* b2  = (const float*)d_in[9];
  const float* W3  = (const float*)d_in[10];
  const float* as3 = (const float*)d_in[11];
  const float* ad3 = (const float*)d_in[12];
  const float* b3  = (const float*)d_in[13];

  const int E = in_sizes[1] / 2;

  LayerBufs B;
  B.E = E; B.ET = E + GN;                 // + self loops
  B.src = ei; B.dst = ei + E;

  // workspace layout (fp32 units)
  float* wsf  = (float*)d_ws;
  B.hbuf      = wsf;                               // [N,128] GEMM output
  B.aggbuf    = B.hbuf + (size_t)GN * GHID;        // [N,128] aggregation
  float* bufC = B.aggbuf + (size_t)GN * GHID;      // [N,128] activated output
  B.asb       = bufC + (size_t)GN * GHID;          // [N,HEADS]
  B.adb       = B.asb + (size_t)GN * GHEADS;
  B.mxb       = (unsigned*)(B.adb + (size_t)GN * GHEADS);
  B.denb      = (float*)(B.mxb + (size_t)GN * GHEADS);
  B.xpk       = (unsigned*)(B.denb + (size_t)GN * GHEADS);  // [N, K/2] packed
  B.wpk       = B.xpk + (size_t)GN * (GIN / 2);             // fragment-packed W

  // layer 1: x[N,256] -> bufC[N,128]
  run_gat_layer<GIN, GHID>(x, W1, as1, ad1, b1, GHEADS, bufC, B, 1, stream);
  // layer 2: bufC[N,128] -> bufC[N,128] (bufC consumed by pack before rewrite)
  run_gat_layer<GHID, GHID>(bufC, W2, as2, ad2, b2, GHEADS, bufC, B, 1, stream);
  // layer 3: bufC[N,128] -> d_out[N,64], heads=1, no activation
  run_gat_layer<GHID, GOUT>(bufC, W3, as3, ad3, b3, 1, (float*)d_out, B, 0, stream);
}
